// MACEDescriptor_9706626089063
// MI455X (gfx1250) — compile-verified, matrix operand order checked by reading, not run on hardware
//
#include <hip/hip_runtime.h>

#define NN 16000
#define EE 256000
#define CHUNK 32000
#define NCHUNK (EE / CHUNK)

typedef __attribute__((ext_vector_type(16))) _Float16 v16h;
typedef __attribute__((ext_vector_type(8)))  _Float16 v8h;
typedef __attribute__((ext_vector_type(8)))  float    v8f;

// ---------------- path tables ----------------
__constant__ int PP_L1[11] = {0,0,0,1,1,1,1,2,2,2,2};
__constant__ int PP_L2[11] = {0,1,2,0,1,1,2,0,1,2,2};
__constant__ int PP_L3[11] = {0,1,2,1,0,2,1,2,1,0,2};

struct EPath { int l1, l2, l3, slot, cg; };
__constant__ EPath EP0[3] = {{0,0,0,0,0},{0,1,1,0,1},{0,2,2,0,2}};
__constant__ EPath EP1[7] = {{0,0,0,0,0},{0,1,1,0,1},{0,2,2,0,2},
                             {1,0,1,1,3},{1,1,0,1,4},{1,1,2,1,5},{1,2,1,2,6}};

// ---------------- CG init (on device, mirrors reference math) ----------------
__device__ double dfact(int n) { double r = 1.0; for (int i = 2; i <= n; ++i) r *= (double)i; return r; }

__device__ void c2r_build(int l, double Ar[5][5], double Ai[5][5]) {
    for (int i = 0; i < 5; ++i) for (int j = 0; j < 5; ++j) { Ar[i][j] = 0.0; Ai[i][j] = 0.0; }
    Ar[l][l] = 1.0;
    double s2 = 0.70710678118654752440;
    for (int m = 1; m <= l; ++m) {
        double sg = (m & 1) ? -1.0 : 1.0;
        Ar[l+m][l+m] = sg * s2;
        Ar[l+m][l-m] = s2;
        Ai[l-m][l+m] = -sg * s2;
        Ai[l-m][l-m] = s2;
    }
}

__global__ void cg_init_kernel(float* __restrict__ cg) {
    int p = threadIdx.x;
    if (p >= 11) return;
    int l1 = PP_L1[p], l2 = PP_L2[p], l3 = PP_L3[p];
    int d1 = 2*l1+1, d2 = 2*l2+1, d3 = 2*l3+1;
    double C[5][5][5];
    for (int i = 0; i < 5; ++i) for (int j = 0; j < 5; ++j) for (int k = 0; k < 5; ++k) C[i][j][k] = 0.0;
    double pre0 = sqrt((double)(2*l3+1) * dfact(l3+l1-l2) * dfact(l3-l1+l2) * dfact(l1+l2-l3) / dfact(l1+l2+l3+1));
    for (int m1 = -l1; m1 <= l1; ++m1)
        for (int m2 = -l2; m2 <= l2; ++m2) {
            int m3 = m1 + m2;
            if (m3 < -l3 || m3 > l3) continue;
            double pre = pre0 * sqrt(dfact(l3+m3)*dfact(l3-m3)*dfact(l1-m1)*dfact(l1+m1)*dfact(l2-m2)*dfact(l2+m2));
            double s = 0.0;
            for (int k = 0; k <= l1 + l2 - l3; ++k) {
                int e0 = k, e1 = l1+l2-l3-k, e2 = l1-m1-k, e3 = l2+m2-k, e4 = l3-l2+m1+k, e5 = l3-l1-m2+k;
                if (e0 < 0 || e1 < 0 || e2 < 0 || e3 < 0 || e4 < 0 || e5 < 0) continue;
                double den = dfact(e0)*dfact(e1)*dfact(e2)*dfact(e3)*dfact(e4)*dfact(e5);
                s += ((k & 1) ? -1.0 : 1.0) / den;
            }
            C[m1+l1][m2+l2][m3+l3] = pre * s;
        }
    double A1r[5][5], A1i[5][5], A2r[5][5], A2i[5][5], A3r[5][5], A3i[5][5];
    c2r_build(l1, A1r, A1i); c2r_build(l2, A2r, A2i); c2r_build(l3, A3r, A3i);
    double Rr[5][5][5], Ri[5][5][5];
    for (int a = 0; a < d1; ++a)
        for (int b = 0; b < d2; ++b)
            for (int c = 0; c < d3; ++c) {
                double sr = 0.0, si = 0.0;
                for (int i = 0; i < d1; ++i)
                    for (int j = 0; j < d2; ++j)
                        for (int k = 0; k < d3; ++k) {
                            double t = C[i][j][k];
                            if (t == 0.0) continue;
                            double ur = A1r[a][i], ui = -A1i[a][i];
                            double vr = A2r[b][j], vi = -A2i[b][j];
                            double wr = A3r[c][k], wi =  A3i[c][k];
                            double uvr = ur*vr - ui*vi, uvi = ur*vi + ui*vr;
                            double zr = uvr*wr - uvi*wi, zi = uvr*wi + uvi*wr;
                            sr += zr * t; si += zi * t;
                        }
                Rr[a][b][c] = sr; Ri[a][b][c] = si;
            }
    double sumr = 0.0, sumi = 0.0;
    for (int a = 0; a < d1; ++a) for (int b = 0; b < d2; ++b) for (int c = 0; c < d3; ++c) {
        sumr += fabs(Rr[a][b][c]); sumi += fabs(Ri[a][b][c]);
    }
    bool useR = (sumr >= sumi);
    double nn = 0.0;
    for (int a = 0; a < d1; ++a) for (int b = 0; b < d2; ++b) for (int c = 0; c < d3; ++c) {
        double v = useR ? Rr[a][b][c] : Ri[a][b][c]; nn += v * v;
    }
    double scale = sqrt((double)(2*l3+1)) / sqrt(nn);
    for (int a = 0; a < d1; ++a) for (int b = 0; b < d2; ++b) for (int c = 0; c < d3; ++c) {
        double v = useR ? Rr[a][b][c] : Ri[a][b][c];
        cg[p*128 + (a*d2 + b)*d3 + c] = (float)(v * scale);
    }
}

// ---------------- WMMA helpers ----------------
__device__ inline v8f wmma16(v16h a, v16h b, v8f c) {
    return __builtin_amdgcn_wmma_f32_16x16x32_f16(false, a, false, b, (short)0, c, false, false);
}

// Fragment loader from an LDS matrix stored [item][K] row-major (f16, 16B-aligned rows).
// item = A-row (M) or B-column (N); per the gfx1250 16-bit layout:
// lanes 0-15: halves 0..7 -> K=kk+0..7, halves 8..15 -> K=kk+16..23
// lanes 16-31: +8 on both K groups.
__device__ inline v16h load_frag(const _Float16* base, int item, int ldK, int kk, int koff) {
    const _Float16* rp = base + (size_t)item * ldK + kk + koff;
    v8h lo = *(const v8h*)(rp);
    v8h hi = *(const v8h*)(rp + 16);
    v16h r;
#pragma unroll
    for (int h = 0; h < 8; ++h) { r[h] = lo[h]; r[8+h] = hi[h]; }
    return r;
}

__device__ inline float siluf(float x) { return x / (1.0f + __expf(-x)); }

// ---------------- generic GEMM: out[M,64] = A[M,K] @ W[K,64] (+ add) ----------------
__global__ __launch_bounds__(128) void gemm64_kernel(const float* __restrict__ A,
        const float* __restrict__ W, const float* __restrict__ add,
        float* __restrict__ out, int M, int K) {
    __shared__ alignas(16) _Float16 sA[64 * 192];
    __shared__ alignas(16) _Float16 sWt[64 * 192];   // transposed: [col][K]
    int tid = threadIdx.x, wave = tid >> 5, lane = tid & 31;
    int nl = lane & 15;
    int koff = (lane < 16) ? 0 : 8;
    int mofs = (lane < 16) ? 0 : 8;
    size_t rowBase = (size_t)blockIdx.x * 64;
    for (int idx = tid; idx < 64 * K; idx += 128) {
        int r = idx / K, c = idx - r * K;
        size_t gr = rowBase + r;
        sA[idx] = (_Float16)((gr < (size_t)M) ? A[gr * K + c] : 0.0f);
    }
    for (int idx = tid; idx < 64 * K; idx += 128) {
        int c = idx / K, k = idx - c * K;
        sWt[idx] = (_Float16)W[(size_t)k * 64 + c];
    }
    __syncthreads();
    v8f vz = {};
    v8f acc[4];
#pragma unroll
    for (int nt = 0; nt < 4; ++nt) acc[nt] = vz;
    for (int kk = 0; kk < K; kk += 32) {
        v16h a = load_frag(sA, wave * 16 + nl, K, kk, koff);
#pragma unroll
        for (int nt = 0; nt < 4; ++nt) {
            v16h bf = load_frag(sWt, nt * 16 + nl, K, kk, koff);
            acc[nt] = wmma16(a, bf, acc[nt]);
        }
    }
#pragma unroll
    for (int nt = 0; nt < 4; ++nt)
#pragma unroll
        for (int r = 0; r < 8; ++r) {
            size_t grow = rowBase + wave * 16 + mofs + r;
            if (grow < (size_t)M) {
                int col = nt * 16 + nl;
                float v = acc[nt][r];
                if (add) v += add[grow * 64 + col];
                out[grow * 64 + col] = v;
            }
        }
}

// ---------------- fused radial MLP: [E,8] ->silu64 ->silu64 ->silu64 -> [E, P*64] ----------------
__global__ __launch_bounds__(128) void radial_kernel(const float* __restrict__ rad,
        const float* __restrict__ W0, const float* __restrict__ W1,
        const float* __restrict__ W2, const float* __restrict__ Wo,
        float* __restrict__ wOut, int nE, int P) {
    __shared__ alignas(16) _Float16 sX[64 * 64];
    __shared__ alignas(16) _Float16 sW[64 * 64];     // transposed [col][K]
    int tid = threadIdx.x, wave = tid >> 5, lane = tid & 31;
    int nl = lane & 15;
    int koff = (lane < 16) ? 0 : 8;
    int mofs = (lane < 16) ? 0 : 8;
    int rowBase = blockIdx.x * 64;
    for (int idx = tid; idx < 64 * 64; idx += 128) {
        int r = idx >> 6, c = idx & 63;
        float v = 0.0f;
        if (c < 8 && rowBase + r < nE) v = rad[(size_t)(rowBase + r) * 8 + c];
        sX[idx] = (_Float16)v;
    }
    for (int idx = tid; idx < 64 * 64; idx += 128) {
        int c = idx >> 6, k = idx & 63;
        sW[idx] = (_Float16)((k < 8) ? W0[k * 64 + c] : 0.0f);
    }
    __syncthreads();
    v8f vz = {};
    v8f acc[4];
    // layer 1 (K=32, zero padded)
#pragma unroll
    for (int nt = 0; nt < 4; ++nt) acc[nt] = vz;
    {
        v16h a = load_frag(sX, wave * 16 + nl, 64, 0, koff);
#pragma unroll
        for (int nt = 0; nt < 4; ++nt)
            acc[nt] = wmma16(a, load_frag(sW, nt * 16 + nl, 64, 0, koff), acc[nt]);
    }
#pragma unroll
    for (int nt = 0; nt < 4; ++nt)
#pragma unroll
        for (int r = 0; r < 8; ++r)
            sX[(wave * 16 + mofs + r) * 64 + nt * 16 + nl] = (_Float16)siluf(acc[nt][r]);
    // layers 2 and 3 (K=64)
    for (int layer = 0; layer < 2; ++layer) {
        const float* Wl = layer ? W2 : W1;
        __syncthreads();
        for (int idx = tid; idx < 64 * 64; idx += 128) {
            int c = idx >> 6, k = idx & 63;
            sW[idx] = (_Float16)Wl[k * 64 + c];
        }
        __syncthreads();
#pragma unroll
        for (int nt = 0; nt < 4; ++nt) acc[nt] = vz;
        for (int kk = 0; kk < 64; kk += 32) {
            v16h a = load_frag(sX, wave * 16 + nl, 64, kk, koff);
#pragma unroll
            for (int nt = 0; nt < 4; ++nt)
                acc[nt] = wmma16(a, load_frag(sW, nt * 16 + nl, 64, kk, koff), acc[nt]);
        }
        __syncthreads();
#pragma unroll
        for (int nt = 0; nt < 4; ++nt)
#pragma unroll
            for (int r = 0; r < 8; ++r)
                sX[(wave * 16 + mofs + r) * 64 + nt * 16 + nl] = (_Float16)siluf(acc[nt][r]);
    }
    // output layer: for each path group, 64 columns of Wo
    for (int g = 0; g < P; ++g) {
        __syncthreads();
        for (int idx = tid; idx < 64 * 64; idx += 128) {
            int c = idx >> 6, k = idx & 63;
            sW[idx] = (_Float16)Wo[(size_t)k * (P * 64) + g * 64 + c];
        }
        __syncthreads();
#pragma unroll
        for (int nt = 0; nt < 4; ++nt) acc[nt] = vz;
        for (int kk = 0; kk < 64; kk += 32) {
            v16h a = load_frag(sX, wave * 16 + nl, 64, kk, koff);
#pragma unroll
            for (int nt = 0; nt < 4; ++nt)
                acc[nt] = wmma16(a, load_frag(sW, nt * 16 + nl, 64, kk, koff), acc[nt]);
        }
#pragma unroll
        for (int nt = 0; nt < 4; ++nt)
#pragma unroll
            for (int r = 0; r < 8; ++r) {
                int grow = rowBase + wave * 16 + mofs + r;
                if (grow < nE)
                    wOut[(size_t)grow * (P * 64) + g * 64 + nt * 16 + nl] = acc[nt][r];
            }
    }
}

// ---------------- element-dependent skip: out[(n,i),c] = sum_k x[(n,i),k] * W[z][k][c] ----------------
__global__ __launch_bounds__(64) void sc_kernel(const float* __restrict__ x,
        const float* __restrict__ W, const int* __restrict__ elem,
        float* __restrict__ out, int dl) {
    int row = blockIdx.x, c = threadIdx.x;
    int n = row / dl;
    int z = elem[n];
    __shared__ float sx[64];
    sx[c] = x[(size_t)row * 64 + c];
    __syncthreads();
    const float* Wz = W + (size_t)z * 4096;
    float acc = 0.0f;
#pragma unroll 8
    for (int k = 0; k < 64; ++k) acc += sx[k] * Wz[k * 64 + c];
    out[(size_t)row * 64 + c] = acc;
}

// ---------------- edge tensor product + scatter ----------------
__global__ __launch_bounds__(256) void edge_kernel(const float* __restrict__ y0,
        const float* __restrict__ y1, const float* __restrict__ sh,
        const int* __restrict__ senders, const int* __restrict__ receivers,
        const float* __restrict__ w, const float* __restrict__ cg,
        float* __restrict__ agg0, float* __restrict__ agg1, float* __restrict__ agg2,
        int blk, int P0, int P1, int P2, int chunkStart, int chunkSize) {
    int k = threadIdx.x;                         // channel 0..63
    int eo = blockIdx.x * blockDim.y + threadIdx.y;
    if (eo >= chunkSize) return;
    int e = chunkStart + eo;
    int s = senders[e];
    int rcv = receivers[e];
    float shv[9];
#pragma unroll
    for (int i = 0; i < 9; ++i) shv[i] = sh[(size_t)e * 9 + i];
    float yv[4];
    yv[0] = y0[(size_t)s * 64 + k];
    if (blk) {
#pragma unroll
        for (int i = 0; i < 3; ++i) yv[1 + i] = y1[((size_t)s * 3 + i) * 64 + k];
    } else {
        yv[1] = yv[2] = yv[3] = 0.0f;
    }
    const int shoff[3] = {0, 1, 4};
    int np = blk ? 7 : 3;
    for (int p = 0; p < np; ++p) {
        EPath q = blk ? EP1[p] : EP0[p];
        int d1 = 2 * q.l1 + 1, d2 = 2 * q.l2 + 1, d3 = 2 * q.l3 + 1;
        const float* cgt = cg + q.cg * 128;
        const float* yy = &yv[(q.l1 == 0) ? 0 : 1];
        const float* ss = &shv[shoff[q.l2]];
        float wv = w[((size_t)eo * np + p) * 64 + k] * 0.0625f;  // / AVG_NN
        float* agg; int P;
        if (q.l3 == 0)      { agg = agg0; P = P0; }
        else if (q.l3 == 1) { agg = agg1; P = P1; }
        else                { agg = agg2; P = P2; }
        for (int m = 0; m < d3; ++m) {
            float acc = 0.0f;
            for (int i = 0; i < d1; ++i)
                for (int j = 0; j < d2; ++j)
                    acc += yy[i] * ss[j] * cgt[(i * d2 + j) * d3 + m];
            atomicAdd(&agg[(((size_t)rcv * d3 + m) * P + q.slot) * 64 + k], acc * wv);
        }
    }
}

// ---------------- correlation-3 product basis (per node, per channel) ----------------
__global__ __launch_bounds__(64) void product_kernel(const float* __restrict__ A0,
        const float* __restrict__ A1, const float* __restrict__ A2,
        const int* __restrict__ elem, const float* __restrict__ w2,
        const float* __restrict__ w3, const float* __restrict__ wnu,
        const float* __restrict__ cg, float* __restrict__ B0, float* __restrict__ B1) {
    int n = blockIdx.x, k = threadIdx.x;
    int z = elem[n];
    float A[9], T[9], Tn[9];
    A[0] = A0[(size_t)n * 64 + k];
#pragma unroll
    for (int i = 0; i < 3; ++i) A[1 + i] = A1[((size_t)n * 3 + i) * 64 + k];
#pragma unroll
    for (int i = 0; i < 5; ++i) A[4 + i] = A2[((size_t)n * 5 + i) * 64 + k];
#pragma unroll
    for (int t = 0; t < 9; ++t) T[t] = A[t];
    const int off[3] = {0, 1, 4};
    float b0 = wnu[(((size_t)z * 3 + 0) * 2 + 0) * 64 + k] * T[0];
    float b1[3];
    {
        float c1 = wnu[(((size_t)z * 3 + 0) * 2 + 1) * 64 + k];
#pragma unroll
        for (int i = 0; i < 3; ++i) b1[i] = c1 * T[1 + i];
    }
    for (int it = 0; it < 2; ++it) {
        const float* wp = it ? w3 : w2;
#pragma unroll
        for (int t = 0; t < 9; ++t) Tn[t] = 0.0f;
        for (int p = 0; p < 11; ++p) {
            int l1 = PP_L1[p], l2 = PP_L2[p], l3 = PP_L3[p];
            int d1 = 2 * l1 + 1, d2 = 2 * l2 + 1, d3 = 2 * l3 + 1;
            const float* cgt = cg + p * 128;
            float c = wp[(((size_t)z * 11) + p) * 64 + k];
            for (int m = 0; m < d3; ++m) {
                float acc = 0.0f;
                for (int i = 0; i < d1; ++i)
                    for (int j = 0; j < d2; ++j)
                        acc += T[off[l1] + i] * A[off[l2] + j] * cgt[(i * d2 + j) * d3 + m];
                Tn[off[l3] + m] += c * acc;
            }
        }
#pragma unroll
        for (int t = 0; t < 9; ++t) T[t] = Tn[t];
        float wn0 = wnu[(((size_t)z * 3 + (it + 1)) * 2 + 0) * 64 + k];
        float wn1 = wnu[(((size_t)z * 3 + (it + 1)) * 2 + 1) * 64 + k];
        b0 += wn0 * T[0];
#pragma unroll
        for (int i = 0; i < 3; ++i) b1[i] += wn1 * T[1 + i];
    }
    B0[(size_t)n * 64 + k] = b0;
#pragma unroll
    for (int i = 0; i < 3; ++i) B1[((size_t)n * 3 + i) * 64 + k] = b1[i];
}

// ---------------- pack features into output [N,512], channel-major for l1 ----------------
__global__ __launch_bounds__(256) void pack_kernel(const float* __restrict__ x0,
        const float* __restrict__ x1, float* __restrict__ out, int b) {
    int n = blockIdx.x, t = threadIdx.x;
    size_t base = (size_t)n * 512 + (size_t)b * 256;
    if (t < 64) {
        out[base + t] = x0[(size_t)n * 64 + t];
    } else {
        int tt = t - 64;          // 0..191 ; col = k*3 + i
        int k = tt / 3, i = tt - k * 3;
        out[base + 64 + tt] = x1[((size_t)n * 3 + i) * 64 + k];
    }
}

// ---------------- host ----------------
extern "C" void kernel_launch(void* const* d_in, const int* in_sizes, int n_in,
                              void* d_out, int out_size, void* d_ws, size_t ws_size,
                              hipStream_t stream) {
    (void)in_sizes; (void)n_in; (void)out_size; (void)ws_size;
    const float* node_feats  = (const float*)d_in[0];
    const int*   node_elem   = (const int*)d_in[1];
    const float* edge_sh     = (const float*)d_in[2];
    const float* edge_radial = (const float*)d_in[3];
    const int*   senders     = (const int*)d_in[4];
    const int*   receivers   = (const int*)d_in[5];
    float* out = (float*)d_out;

    float* p   = (float*)d_ws;
    float* cg  = p; p += 2048;
    float* x0  = p; p += (size_t)NN * 64;
    float* x1  = p; p += (size_t)NN * 192;
    float* y0b = p; p += (size_t)NN * 64;
    float* y1b = p; p += (size_t)NN * 192;
    float* sc0b = p; p += (size_t)NN * 64;
    float* sc1b = p; p += (size_t)NN * 192;
    float* agg0 = p; p += (size_t)NN * 2 * 64;     // [N][1][P0<=2][64]
    float* agg1 = p; p += (size_t)NN * 9 * 64;     // [N][3][P1<=3][64]
    float* agg2 = p; p += (size_t)NN * 10 * 64;    // [N][5][P2<=2][64]
    float* A0  = p; p += (size_t)NN * 64;
    float* A1  = p; p += (size_t)NN * 3 * 64;
    float* A2  = p; p += (size_t)NN * 5 * 64;
    float* B0  = p; p += (size_t)NN * 64;
    float* B1  = p; p += (size_t)NN * 3 * 64;
    float* wbuf = p; p += (size_t)CHUNK * 7 * 64;

    cg_init_kernel<<<1, 32, 0, stream>>>(cg);

    for (int b = 0; b < 2; ++b) {
        const float* upL0  = (const float*)d_in[b ? 20 : 6];
        const float* upL1  = b ? (const float*)d_in[21] : nullptr;
        const float* rw0   = (const float*)d_in[b ? 22 : 7];
        const float* rw1   = (const float*)d_in[b ? 23 : 8];
        const float* rw2   = (const float*)d_in[b ? 24 : 9];
        const float* rwo   = (const float*)d_in[b ? 25 : 10];
        const float* linl0 = (const float*)d_in[b ? 26 : 11];
        const float* linl1 = (const float*)d_in[b ? 27 : 12];
        const float* linl2 = (const float*)d_in[b ? 28 : 13];
        const float* scl0  = (const float*)d_in[b ? 29 : 14];
        const float* scl1  = b ? (const float*)d_in[30] : nullptr;
        const float* pw2   = (const float*)d_in[b ? 31 : 15];
        const float* pw3   = (const float*)d_in[b ? 32 : 16];
        const float* pwnu  = (const float*)d_in[b ? 33 : 17];
        const float* plin0 = (const float*)d_in[b ? 34 : 18];
        const float* plin1 = (const float*)d_in[b ? 35 : 19];
        int P0 = b ? 2 : 1, P1 = b ? 3 : 1, P2 = b ? 2 : 1;
        int np = b ? 7 : 3;
        const float* xin0 = b ? x0 : node_feats;

        // linear_up (WMMA) + element-dependent skip
        gemm64_kernel<<<250, 128, 0, stream>>>(xin0, upL0, nullptr, y0b, NN, 64);
        if (b) gemm64_kernel<<<750, 128, 0, stream>>>(x1, upL1, nullptr, y1b, NN * 3, 64);
        sc_kernel<<<NN, 64, 0, stream>>>(xin0, scl0, node_elem, sc0b, 1);
        if (b) sc_kernel<<<NN * 3, 64, 0, stream>>>(x1, scl1, node_elem, sc1b, 3);

        hipMemsetAsync(agg0, 0, (size_t)NN * 2 * 64 * sizeof(float), stream);
        hipMemsetAsync(agg1, 0, (size_t)NN * 9 * 64 * sizeof(float), stream);
        hipMemsetAsync(agg2, 0, (size_t)NN * 10 * 64 * sizeof(float), stream);

        // chunked radial MLP (WMMA) + edge tensor-product scatter
        for (int c = 0; c < NCHUNK; ++c) {
            int cs = c * CHUNK;
            radial_kernel<<<CHUNK / 64, 128, 0, stream>>>(edge_radial + (size_t)cs * 8,
                rw0, rw1, rw2, rwo, wbuf, CHUNK, np);
            edge_kernel<<<dim3(CHUNK / 4), dim3(64, 4), 0, stream>>>(y0b, y1b, edge_sh,
                senders, receivers, wbuf, cg, agg0, agg1, agg2,
                b, P0, P1, P2, cs, CHUNK);
        }

        // post-aggregation linears (WMMA over (path,channel) K axis)
        gemm64_kernel<<<250, 128, 0, stream>>>(agg0, linl0, nullptr, A0, NN, P0 * 64);
        gemm64_kernel<<<750, 128, 0, stream>>>(agg1, linl1, nullptr, A1, NN * 3, P1 * 64);
        gemm64_kernel<<<1250, 128, 0, stream>>>(agg2, linl2, nullptr, A2, NN * 5, P2 * 64);

        // equivariant product basis (correlation 3)
        product_kernel<<<NN, 64, 0, stream>>>(A0, A1, A2, node_elem, pw2, pw3, pwnu, cg, B0, B1);

        // final linear + skip (fused add epilogue)
        gemm64_kernel<<<250, 128, 0, stream>>>(B0, plin0, sc0b, x0, NN, 64);
        gemm64_kernel<<<750, 128, 0, stream>>>(B1, plin1, b ? sc1b : nullptr, x1, NN * 3, 64);

        pack_kernel<<<NN, 256, 0, stream>>>(x0, x1, out, b);
    }
}
